// TreeCompositionGRU_69793218560553
// MI455X (gfx1250) — matrix operand order repaired and checked by reading, not compiled, not run on hardware
//
#include <hip/hip_runtime.h>
#include <hip/hip_bf16.h>

typedef __attribute__((ext_vector_type(16))) _Float16 v16h;
typedef __attribute__((ext_vector_type(8)))  float    v8f;

#define B_SZ   256
#define NNODE  1023
#define WD     300
#define TD     50
#define RD     50
#define HID    150
// K layout: x-part 550 -> padded 576 (18 k-tiles of 32), h-part 150 -> 160 (5 k-tiles)
#define KPAD   736
#define KT     23
#define XKT    18
// N layout: rz combined 300->304 (19 tiles), i_n 150->160 (10), h_n 150->160 (10)
#define NT     39
#define NPAD   624
#define COL_IN 304
#define COL_HN 464

#define ROWS   32                  // rows per workgroup (2 WMMA row-tiles)
#define NTHR   256                 // 8 waves
// dynamic LDS partition (bytes)
#define LDS_A_BYTES (2 * KPAD * 16 * 2)          // 47104
#define LDS_C_OFF   LDS_A_BYTES
#define LDS_BYTES   (LDS_A_BYTES + 2 * 16 * NPAD * 4)   // 126976

// ---------------- workspace offsets (bytes) ----------------
#define OFF_W    0                  // packed f16 weights: NT*KT*32*16 halves = 918528 B
#define OFF_BIAS 918528             // 624 f32
#define OFF_H    921088             // f32 recurrent h: 256*256*150*4 = 39321600 B
#define OFF_S0   40242688           // f16 states ping: 256*256*150*2
#define OFF_S1   59903488           // f16 states pong

// Pack W_ih(450x550) + W_hh(450x150) into WMMA-B lane layout, combined K.
// B tile layout used here (contiguous k, same as A): lane = (kLocal/16)*16 + n ; element e = kLocal%16.
__global__ __launch_bounds__(128) void gru_pack_weights(
    const float* __restrict__ W_ih, const float* __restrict__ W_hh,
    const float* __restrict__ b_ih, const float* __restrict__ b_hh,
    _Float16* __restrict__ Wpack, float* __restrict__ biasPack) {
  int tid = blockIdx.x * blockDim.x + threadIdx.x;
  if (tid < NPAD) {
    float bv = 0.f;
    if (tid < 300)                               bv = b_ih[tid] + b_hh[tid];
    else if (tid >= COL_IN && tid < COL_IN+HID)  bv = b_ih[300 + tid - COL_IN];
    else if (tid >= COL_HN && tid < COL_HN+HID)  bv = b_hh[300 + tid - COL_HN];
    biasPack[tid] = bv;
  }
  if (tid >= NT * KT * 32) return;
  const int lane   = tid & 31;
  const int rem    = tid >> 5;
  const int kt     = rem % KT;
  const int t      = rem / KT;
  const int n      = lane & 15;
  const int laneHi = lane >> 4;
  const int c      = t * 16 + n;
  _Float16* dst = Wpack + ((size_t)(t * KT + kt) * 32 + lane) * 16;
#pragma unroll
  for (int e = 0; e < 16; ++e) {
    const int k = kt * 32 + laneHi * 16 + e;
    float v = 0.f;
    if (c < 300) {                                   // r,z combined columns
      if (k < 550)                   v = W_ih[c * 550 + k];
      else if (k >= 576 && k < 726)  v = W_hh[c * 150 + (k - 576)];
    } else if (c >= COL_IN && c < COL_IN + HID) {    // i_n columns (x-part only)
      const int g = 300 + (c - COL_IN);
      if (k < 550)                   v = W_ih[g * 550 + k];
    } else if (c >= COL_HN && c < COL_HN + HID) {    // h_n columns (h-part only)
      const int g = 300 + (c - COL_HN);
      if (k >= 576 && k < 726)       v = W_hh[g * 150 + (k - 576)];
    }
    dst[e] = (_Float16)v;
  }
}

// One fused GRU step for 32 (batch,parent) rows per workgroup (2 row-tiles).
// 8 waves; gather+concat x||h into LDS (WMMA-A swizzle), each B fragment feeds
// two v_wmma_f32_16x16x32_f16 (one per row-tile), C staged in LDS, fused
// sigmoid/tanh GRU update.
__global__ __launch_bounds__(NTHR) void gru_phase(
    const float* __restrict__ word, const float* __restrict__ tag,
    const float* __restrict__ rel,  const float* __restrict__ rel_self,
    const _Float16* __restrict__ Wpack, const float* __restrict__ biasPack,
    float* __restrict__ hbuf, const _Float16* __restrict__ stateIn,
    _Float16* __restrict__ stateOut, float* __restrict__ outp,
    int lvl, int step, int M) {
  extern __shared__ __align__(32) char smem[];
  _Float16* ldsA = (_Float16*)smem;                 // [2][KPAD*16]
  float*    ldsC = (float*)(smem + LDS_C_OFF);      // [2][16][NPAD]

  const int tid     = threadIdx.x;
  const int rowBase = blockIdx.x * ROWS;

  // ---------------- build A tiles (x concat || h) in WMMA-A swizzle ---------
  {
    const int m   = tid >> 3;       // local row 0..31
    const int seg = tid & 7;        // 8 threads per row, 92 features each
    const int at  = m >> 4;         // which A row-tile
    const int mL  = m & 15;
    const int row = rowBase + m;
    const int b   = row / M;
    const int p   = row - b * M;
    const int node = (1 << lvl) - 1 + p;
    const int nidx = (step == 0) ? (2 * node + 1) : (step == 1) ? node : (2 * node + 2);
    const size_t eb = (size_t)b * NNODE + nidx;
    const float* wp = word + eb * WD;
    const float* tp = tag  + eb * TD;
    const float* rp = (step == 1) ? rel_self : (rel + eb * RD);
    const int cL = 2 * p + ((step == 2) ? 1 : 0);
    const _Float16* kp = (lvl < 8 && step != 1)
                         ? (stateIn + ((size_t)(b * (2 * M) + cL)) * HID) : nullptr;
    const float* hp = (step > 0) ? (hbuf + (size_t)row * HID) : nullptr;
    _Float16* aBase = ldsA + (size_t)at * (KPAD * 16);
    const int f0 = seg * 92;
#pragma unroll 4
    for (int f = f0; f < f0 + 92; ++f) {
      float v = 0.f;
      if      (f < WD)                 v = wp[f];
      else if (f < 350)                v = tp[f - WD];
      else if (f < 400)                v = rp[f - 350];
      else if (f < 550)                v = kp ? (float)kp[f - 400] : 0.f;
      else if (f >= 576 && f < 726)    v = hp ? hp[f - 576] : 0.f;
      // A k order used here (contiguous, same as B): K = laneHi*16 + e
      const int kt = f >> 5, kl = f & 31;
      const int laneHi = kl >> 4;
      const int e = kl & 15;
      aBase[((kt * 32) + laneHi * 16 + mL) * 16 + e] = (_Float16)v;
    }
  }
  __syncthreads();

  // ---------------- WMMA GEMM: C(32xNPAD) = A(32xKPAD) * Wpack -------------
  {
    const int lane = tid & 31;
    const int wave = __builtin_amdgcn_readfirstlane(tid >> 5);  // scalar tile loop
    for (int t = wave; t < NT; t += 8) {
      int kt0 = 0, kt1 = KT;
      if (t >= 29)      { kt0 = XKT; kt1 = KT;  }    // h_n: h-part K only
      else if (t >= 19) { kt0 = 0;   kt1 = XKT; }    // i_n: x-part K only
      v8f acc0 = {0.f, 0.f, 0.f, 0.f, 0.f, 0.f, 0.f, 0.f};
      v8f acc1 = {0.f, 0.f, 0.f, 0.f, 0.f, 0.f, 0.f, 0.f};
      const _Float16* bp = Wpack + ((size_t)(t * KT + kt0) * 32 + lane) * 16;
      const _Float16* ap = ldsA + (size_t)(kt0 * 32 + lane) * 16;
      if (t + 8 < NT)   // prefetch next tile's B stream (global_prefetch_b8)
        __builtin_prefetch(Wpack + ((size_t)((t + 8) * KT) * 32 + lane) * 16, 0, 1);
      for (int kt = kt0; kt < kt1; ++kt) {
        v16h b  = *(const v16h*)bp;
        v16h a0 = *(const v16h*)ap;
        v16h a1 = *(const v16h*)(ap + KPAD * 16);
        bp += 32 * 16;
        ap += 32 * 16;
        acc0 = __builtin_amdgcn_wmma_f32_16x16x32_f16(
            false, a0, false, b, (short)0, acc0, false, false);
        acc1 = __builtin_amdgcn_wmma_f32_16x16x32_f16(
            false, a1, false, b, (short)0, acc1, false, false);
      }
      const int n = lane & 15, laneHi = lane >> 4;
#pragma unroll
      for (int r = 0; r < 8; ++r) {
        ldsC[(r + laneHi * 8) * NPAD + t * 16 + n] = acc0[r];
        ldsC[(16 * NPAD) + (r + laneHi * 8) * NPAD + t * 16 + n] = acc1[r];
      }
    }
  }
  __syncthreads();

  // ---------------- fused GRU pointwise update -----------------------------
  for (int idx = tid; idx < ROWS * HID; idx += NTHR) {
    const int m = idx / HID;
    const int j = idx - m * HID;
    const float* C = ldsC + (size_t)(m >> 4) * (16 * NPAD) + (m & 15) * NPAD;
    const float rv  = C[j]            + biasPack[j];
    const float zv  = C[HID + j]      + biasPack[HID + j];
    const float inv = C[COL_IN + j]   + biasPack[COL_IN + j];
    const float hnv = C[COL_HN + j]   + biasPack[COL_HN + j];
    const float r = 1.f / (1.f + expf(-rv));
    const float z = 1.f / (1.f + expf(-zv));
    const float n = tanhf(inv + r * hnv);
    const int row = rowBase + m;
    const float hprev = (step == 0) ? 0.f : hbuf[(size_t)row * HID + j];
    const float hnew = (1.f - z) * n + z * hprev;
    hbuf[(size_t)row * HID + j] = hnew;
    if (step == 2) {
      stateOut[(size_t)row * HID + j] = (_Float16)hnew;
      if (lvl == 0) outp[(size_t)row * HID + j] = hnew;   // M==1 -> row==b
    }
  }
}

extern "C" void kernel_launch(void* const* d_in, const int* in_sizes, int n_in,
                              void* d_out, int out_size, void* d_ws, size_t ws_size,
                              hipStream_t stream) {
  const float* word     = (const float*)d_in[0];
  const float* tag      = (const float*)d_in[1];
  const float* rel      = (const float*)d_in[2];
  const float* rel_self = (const float*)d_in[3];
  const float* W_ih     = (const float*)d_in[4];
  const float* W_hh     = (const float*)d_in[5];
  const float* b_ih     = (const float*)d_in[6];
  const float* b_hh     = (const float*)d_in[7];
  float* outp = (float*)d_out;

  char* ws = (char*)d_ws;
  _Float16* Wpack    = (_Float16*)(ws + OFF_W);
  float*    biasPack = (float*)   (ws + OFF_BIAS);
  float*    hbuf     = (float*)   (ws + OFF_H);
  _Float16* S0       = (_Float16*)(ws + OFF_S0);
  _Float16* S1       = (_Float16*)(ws + OFF_S1);

  // one-time (per call) weight/bias packing into WMMA-B layout
  {
    const int nThreads = NT * KT * 32;   // 28704
    gru_pack_weights<<<(nThreads + 127) / 128, 128, 0, stream>>>(
        W_ih, W_hh, b_ih, b_hh, Wpack, biasPack);
  }

  // 9 levels x 3 GRU steps, bottom-up; states ping-pong between S0/S1
  for (int lvl = 8; lvl >= 0; --lvl) {
    const int M = 1 << lvl;
    _Float16* sOut      = (lvl & 1) ? S1 : S0;
    const _Float16* sIn = (lvl & 1) ? S0 : S1;   // children = level lvl+1
    for (int step = 0; step < 3; ++step) {
      gru_phase<<<8 * M, NTHR, LDS_BYTES, stream>>>(
          word, tag, rel, rel_self, Wpack, biasPack,
          hbuf, sIn, sOut, outp, lvl, step, M);
    }
  }
}
